// WeightQuantGMMModel_44856638439912
// MI455X (gfx1250) — compile-verified, matrix operand order checked by reading, not run on hardware
//
#include <hip/hip_runtime.h>

typedef __attribute__((ext_vector_type(16))) _Float16 v16h;
typedef __attribute__((ext_vector_type(2)))  __fp16   v2hp;   // cvt_pkrtz result type
typedef __attribute__((ext_vector_type(8)))  float    v8f;

#define M_TOT   2048
#define K_TOT   1024
#define N_TOT   1024
#define G_TOT   64
#define GM      32              // rows per group (M_TOT / G_TOT)
#define BN      256             // N columns per workgroup
#define BK      32              // K depth per iteration
#define NITER   (K_TOT / BK)    // 32
#define NWG_N   (N_TOT / BN)    // 4
#define THREADS 256             // 8 wave32s

// LDS dword layout within one buffer:
//  B fragments: 16 n-tiles * 32 lanes * 8 dwords (v16h per lane) = 4096 dwords
//  A fragments:  2 m-tiles * 32 lanes * 8 dwords                =  512 dwords
#define B_DW    (16 * 32 * 8)
#define A_DW    (2 * 32 * 8)
#define BUF_DW  (B_DW + A_DW)   // 4608 dwords = 18 KB per buffer

// Single v_cvt_pk_rtz_f16_f32 per packed pair -- no bitfield reassembly.
static __device__ __forceinline__ unsigned int pack2h(float a, float b) {
    union { v2hp h; unsigned int u; } cv;
    cv.h = __builtin_amdgcn_cvt_pkrtz(a, b);
    return cv.u;
}

__global__ __launch_bounds__(THREADS)
void wq_gmm_kernel(const float* __restrict__ x,
                   const int* __restrict__ w,
                   const float* __restrict__ scale,
                   const float* __restrict__ offs,
                   const long long* __restrict__ glist,
                   const int* __restrict__ gtype,
                   float* __restrict__ out)
{
    __shared__ __align__(32) unsigned int lds[2 * BUF_DW];
    __shared__ float rowsum[GM];

    const int tid  = threadIdx.x;
    const int lane = tid & 31;
    const int wv   = tid >> 5;          // wave id 0..7

    const int g    = blockIdx.x / NWG_N;
    const int nb   = blockIdx.x % NWG_N;
    const int n_wg = nb * BN;

    // ---- group row range from group_list (cumulative or per-count) ----
    long long rs = 0, re = 0;
    if (gtype[0] == 0) {
        rs = (g > 0) ? glist[g - 1] : 0;
        re = glist[g];
    } else {
        long long acc = 0;
        for (int i = 0; i < g; ++i) acc += glist[i];
        rs = acc; re = acc + glist[g];
    }
    const int row_start = (int)rs;
    const int row_end   = (int)re;

    // ---- f32 row-sums of x for the offset term (exact, matches reference) ----
    if (tid < GM) rowsum[tid] = 0.f;
    __syncthreads();
    {
        const int ml  = tid >> 3;       // 0..31 local row
        const int seg = tid & 7;        // 0..7, 128 floats each
        int row = row_start + ml; if (row >= M_TOT) row = M_TOT - 1;
        const float4* xr = (const float4*)(x + (size_t)row * K_TOT + seg * 128);
        float s = 0.f;
        #pragma unroll
        for (int i = 0; i < 32; ++i) {
            float4 v = xr[i];
            s += v.x + v.y + v.z + v.w;
        }
        atomicAdd(&rowsum[ml], s);      // ds_add_f32
    }

    // ---- register staging for double-buffered pipeline ----
    int4   rw[4][2];
    float2 rx[2];

    auto load_regs = [&](int it) {
        const int k0 = it * BK;
        #pragma unroll
        for (int j = 0; j < 4; ++j) {
            const int idx = tid + THREADS * j;          // 0..1023 units
            const int kp  = (idx >> 6) << 1;            // even k row: 0,2,..,30
            const int nu  = (idx & 63) << 2;            // n offset: 0,4,..,252
            const int* p = w + ((size_t)g * K_TOT + (size_t)(k0 + kp)) * N_TOT
                             + n_wg + nu;
            rw[j][0] = *(const int4*)p;                 // row kp,   4 ints
            rw[j][1] = *(const int4*)(p + N_TOT);       // row kp+1, 4 ints
        }
        #pragma unroll
        for (int j = 0; j < 2; ++j) {
            const int p2  = tid + THREADS * j;          // 0..511 pair units
            const int ml  = p2 >> 4;                    // 0..31 local row
            const int kp2 = (p2 & 15) << 1;             // even k: 0..30
            int row = row_start + ml; if (row >= M_TOT) row = M_TOT - 1;
            rx[j] = *(const float2*)(x + (size_t)row * K_TOT + k0 + kp2);
        }
    };

    // Store converted f16 data into per-lane WMMA fragment layout:
    // 16-bit operand, lane = col + 16*((k>>3)&1), half index h = (k&7) + 8*(k>>4)
    // k even => halves (k, k+1) share dword (h>>1) = ((k&7)>>1) + 4*(k>>4)
    auto store_lds = [&](unsigned int* buf) {
        #pragma unroll
        for (int j = 0; j < 4; ++j) {
            const int idx = tid + THREADS * j;
            const int kp  = (idx >> 6) << 1;
            const int nu  = (idx & 63) << 2;
            const int lk  = (kp >> 3) & 1;
            const int dw  = ((kp & 7) >> 1) + ((kp >> 4) << 2);
            unsigned int wd0 = pack2h((float)rw[j][0].x, (float)rw[j][1].x);
            unsigned int wd1 = pack2h((float)rw[j][0].y, (float)rw[j][1].y);
            unsigned int wd2 = pack2h((float)rw[j][0].z, (float)rw[j][1].z);
            unsigned int wd3 = pack2h((float)rw[j][0].w, (float)rw[j][1].w);
            #pragma unroll
            for (int c = 0; c < 4; ++c) {
                const int n  = nu + c;
                const int tt = n >> 4;                      // n-tile 0..15
                const int ln = (n & 15) + (lk << 4);        // fragment lane
                unsigned int v = (c == 0) ? wd0 : (c == 1) ? wd1 : (c == 2) ? wd2 : wd3;
                buf[(tt * 32 + ln) * 8 + dw] = v;
            }
        }
        #pragma unroll
        for (int j = 0; j < 2; ++j) {
            const int p2  = tid + THREADS * j;
            const int ml  = p2 >> 4;
            const int kp2 = (p2 & 15) << 1;
            const int lnA = (ml & 15) + (((kp2 >> 3) & 1) << 4);
            const int dwA = ((kp2 & 7) >> 1) + ((kp2 >> 4) << 2);
            buf[B_DW + ((ml >> 4) * 32 + lnA) * 8 + dwA] = pack2h(rx[j].x, rx[j].y);
        }
    };

    v8f acc00 = {}, acc01 = {}, acc10 = {}, acc11 = {};
    const int ntA = wv * 2;     // this wave's pair of 16-wide n-tiles

    // ---- pipeline prologue ----
    load_regs(0);
    __syncthreads();            // rowsum atomics done before buffers reused
    store_lds(lds);
    __syncthreads();

    // ---- main K loop: 32 iterations, 4 WMMAs each; unroll 2 => static
    // ping-pong buffers, no per-iter buffer-select SALU ----
    #pragma unroll 2
    for (int it = 0; it < NITER; ++it) {
        unsigned int* cur = lds + (it & 1) * BUF_DW;
        unsigned int* nxt = lds + ((it + 1) & 1) * BUF_DW;

        if (it + 1 < NITER) load_regs(it + 1);      // hide HBM latency under WMMA
        if (it + 2 < NITER) {                       // warm L2: full 32KB K+2 tile
            const int* pf = w + ((size_t)g * K_TOT + (size_t)((it + 2) * BK + (tid >> 3))) * N_TOT
                              + n_wg + ((tid & 7) << 5);
            __builtin_prefetch(pf, 0, 1);           // global_prefetch_b8
        }

        v16h a0 = *(const v16h*)(cur + B_DW + (0 * 32 + lane) * 8);
        v16h a1 = *(const v16h*)(cur + B_DW + (1 * 32 + lane) * 8);
        v16h b0 = *(const v16h*)(cur + ((ntA + 0) * 32 + lane) * 8);
        v16h b1 = *(const v16h*)(cur + ((ntA + 1) * 32 + lane) * 8);

        acc00 = __builtin_amdgcn_wmma_f32_16x16x32_f16(false, a0, false, b0, (short)0, acc00, false, false);
        acc01 = __builtin_amdgcn_wmma_f32_16x16x32_f16(false, a0, false, b1, (short)0, acc01, false, false);
        acc10 = __builtin_amdgcn_wmma_f32_16x16x32_f16(false, a1, false, b0, (short)0, acc10, false, false);
        acc11 = __builtin_amdgcn_wmma_f32_16x16x32_f16(false, a1, false, b1, (short)0, acc11, false, false);

        if (it + 1 < NITER) store_lds(nxt);
        __syncthreads();
    }

    // ---- epilogue: out = scale * (acc + off * rowsum) ----
    // C/D layout: VGPR r, lanes 0-15 -> M=r,   N=lane
    //             VGPR r, lanes16-31 -> M=r+8, N=lane-16
    const int ln15 = lane & 15;
    const int lhi  = (lane >> 4) << 3;  // 0 or 8

    #pragma unroll
    for (int mt = 0; mt < 2; ++mt) {
        #pragma unroll
        for (int nt = 0; nt < 2; ++nt) {
            v8f a = (mt == 0) ? (nt == 0 ? acc00 : acc01)
                              : (nt == 0 ? acc10 : acc11);
            const int n = n_wg + (ntA + nt) * 16 + ln15;
            const float s = scale[(size_t)g * N_TOT + n];
            const float o = offs [(size_t)g * N_TOT + n];
            #pragma unroll
            for (int r = 0; r < 8; ++r) {
                const int m   = mt * 16 + lhi + r;
                const int row = row_start + m;
                if (row < row_end) {
                    out[(size_t)row * N_TOT + n] = s * (a[r] + o * rowsum[m]);
                }
            }
        }
    }
}

extern "C" void kernel_launch(void* const* d_in, const int* in_sizes, int n_in,
                              void* d_out, int out_size, void* d_ws, size_t ws_size,
                              hipStream_t stream) {
    const float*     x  = (const float*)d_in[0];
    const int*       w  = (const int*)d_in[1];
    const float*     sc = (const float*)d_in[2];
    const float*     of = (const float*)d_in[3];
    const long long* gl = (const long long*)d_in[4];
    const int*       gt = (const int*)d_in[5];
    float*           o  = (float*)d_out;

    dim3 grid(G_TOT * NWG_N);   // 256 workgroups
    dim3 block(THREADS);        // 8 wave32s
    wq_gmm_kernel<<<grid, block, 0, stream>>>(x, w, sc, of, gl, gt, o);
}
